// EquivariantTransformereNet_34479997453190
// MI455X (gfx1250) — matrix operand order repaired and checked by reading, not compiled, not run on hardware
//
#include <hip/hip_runtime.h>
#include <hip/hip_bf16.h>
#include <math.h>

#define D_EMBC 8
#define KNNK   20
#define N_RBF  16
#define H_DIM  32
#define BATCH  4
#define NPTS   4096
#define ZF     14      // floats per point in z: pos(3) vel(3) mass(8)
#define CHUNK  64

typedef __attribute__((ext_vector_type(16))) _Float16 v16h;
typedef __attribute__((ext_vector_type(8)))  float    v8f;
typedef __attribute__((ext_vector_type(2)))  float    v2f;

__device__ __forceinline__ float gelu_tanh(float x) {
    float x3 = x * x * x;
    return 0.5f * x * (1.0f + tanhf(0.7978845608028654f * (x + 0.044715f * x3)));
}

// ---------------------------------------------------------------------------
// Kernel 1: conditioning MLP.  t(B) -> timestep emb(32) ++ cond(4) -> 32 -> 32 -> 8
// One block of 128 threads = (4 batches) x (32 feature lanes).
// ---------------------------------------------------------------------------
__global__ void cond_mlp_kernel(const float* __restrict__ t,
                                const float* __restrict__ conditioning,
                                const float* __restrict__ W1, const float* __restrict__ b1,
                                const float* __restrict__ W2, const float* __restrict__ b2,
                                const float* __restrict__ W3, const float* __restrict__ b3,
                                float* __restrict__ cond_out) {
    __shared__ float h0[BATCH][36];
    __shared__ float h1[BATCH][32];
    __shared__ float h2[BATCH][32];
    int tid = threadIdx.x;
    int b = tid >> 5, j = tid & 31;
    float tv = t[b];
    const float neg_ln1e4_over = -logf(10000.0f) / 15.0f;
    if (j < 16) {
        float f = expf(neg_ln1e4_over * (float)j);
        h0[b][j] = sinf(tv * f);
    } else {
        float f = expf(neg_ln1e4_over * (float)(j - 16));
        h0[b][j] = cosf(tv * f);
    }
    if (j < 4) h0[b][32 + j] = conditioning[b * 4 + j];
    __syncthreads();

    {   // layer 1: 36 -> 32
        float acc = b1[j];
        for (int i = 0; i < 36; ++i) acc += h0[b][i] * W1[i * 32 + j];
        h1[b][j] = gelu_tanh(acc);
    }
    __syncthreads();
    {   // layer 2: 32 -> 32
        float acc = b2[j];
        for (int i = 0; i < 32; ++i) acc += h1[b][i] * W2[i * 32 + j];
        h2[b][j] = gelu_tanh(acc);
    }
    __syncthreads();
    if (j < 8) {   // layer 3: 32 -> 8
        float acc = b3[j];
        for (int i = 0; i < 32; ++i) acc += h2[b][i] * W3[i * 8 + j];
        cond_out[b * 8 + j] = acc;
    }
}

// ---------------------------------------------------------------------------
// Kernel 2: KNN via f32 WMMA Gram-matrix tiles.
// One wave (32 threads) per (batch, 16-query-row tile).  grid = B*256.
// d2 = |q|^2 + |c|^2 - 2 q.c ; dot tiles come from V_WMMA_F32_16X16X4_F32.
// Top-20 lists live in LDS; the hot scan keeps only a threshold in a VGPR.
// ---------------------------------------------------------------------------
__global__ void knn_kernel(const float* __restrict__ z, const float* __restrict__ mask,
                           int* __restrict__ knn_idx) {
    int blk = blockIdx.x;
    int b = blk >> 8;
    int rowBase = (blk & 255) << 4;
    int l = threadIdx.x;
    int half = l >> 4;
    int lm = l & 15;

    __shared__ float qpos[16][4];            // {x,y,z,0}
    __shared__ float qnb[16];                // |q|^2 (+1e9 if masked)
    __shared__ float cpos[CHUNK][4];         // {x,y,z,0}
    __shared__ float cnb[CHUNK];             // |c|^2 (+1e9 if masked)
    __shared__ float d2buf[16][CHUNK + 4];   // +4 keeps rows 16B aligned
    __shared__ float md[32][KNNK];           // per-lane sorted top-20 distances
    __shared__ int   mi[32][KNNK];           // per-lane sorted top-20 indices

    const float* zb = z + (size_t)b * NPTS * ZF;
    const float* mb = mask + (size_t)b * NPTS;

    if (l < 16) {
        const float* p = zb + (size_t)(rowBase + l) * ZF;
        float x = p[0], y = p[1], zc = p[2];
        qpos[l][0] = x; qpos[l][1] = y; qpos[l][2] = zc; qpos[l][3] = 0.0f;
        qnb[l] = x * x + y * y + zc * zc + (mb[rowBase + l] <= 0.0f ? 1.0e9f : 0.0f);
    }
    for (int k = 0; k < KNNK; ++k) { md[l][k] = 3.0e38f; mi[l][k] = 0; }
    __syncthreads();

    // A fragment (16x4 f32): lanes0-15 carry K=0,1 = (x,y); lanes16-31 K=2,3 = (z,0)
    v2f afrag = *(const v2f*)&qpos[lm][2 * half];
    float qn8[8];
#pragma unroll
    for (int v = 0; v < 8; ++v) qn8[v] = qnb[v + 8 * half];

    float thr = 3.0e38f;

    for (int base = 0; base < NPTS; base += CHUNK) {
        for (int s = 0; s < 2; ++s) {                     // stage 64 candidates
            int c = l + 32 * s;
            const float* p = zb + (size_t)(base + c) * ZF;
            float x = p[0], y = p[1], zc = p[2];
            cpos[c][0] = x; cpos[c][1] = y; cpos[c][2] = zc; cpos[c][3] = 0.0f;
            cnb[c] = x * x + y * y + zc * zc + (mb[base + c] <= 0.0f ? 1.0e9f : 0.0f);
        }
        __syncthreads();

        for (int tt = 0; tt < CHUNK / 16; ++tt) {         // 4 WMMA tiles per chunk
            int n = tt * 16 + lm;
            v2f bfrag = *(const v2f*)&cpos[n][2 * half];  // branchless b64 LDS load
            v8f czero = {};
            v8f dots = __builtin_amdgcn_wmma_f32_16x16x4_f32(false, afrag, false, bfrag,
                                                             (short)0, czero, false, false);
            float cnv = cnb[n];
#pragma unroll
            for (int v = 0; v < 8; ++v)
                d2buf[v + 8 * half][tt * 16 + lm] = (qn8[v] + cnv) - 2.0f * dots[v];
        }
        __syncthreads();

        // selection: lane owns query row lm; half selects candidate sub-range of 32
        const float4* rowp = (const float4*)&d2buf[lm][32 * half];
        for (int q4 = 0; q4 < 8; ++q4) {
            float4 dv = rowp[q4];
            float dd[4] = { dv.x, dv.y, dv.z, dv.w };
#pragma unroll
            for (int s = 0; s < 4; ++s) {
                float d = dd[s];
                if (d < thr) {                            // rare path: LDS list insert
                    int id = base + 32 * half + q4 * 4 + s;
                    int k = KNNK - 1;
                    while (k > 0 && md[l][k - 1] > d) {
                        md[l][k] = md[l][k - 1];
                        mi[l][k] = mi[l][k - 1];
                        --k;
                    }
                    md[l][k] = d;
                    mi[l][k] = id;
                    thr = md[l][KNNK - 1];
                }
            }
        }
        __syncthreads();
    }

    // merge the two partial lists per row (tie -> smaller index, matches top_k)
    if (l < 16) {
        int i0 = 0, i1 = 0;
        int* o = knn_idx + ((size_t)b * NPTS + rowBase + l) * KNNK;
        for (int k = 0; k < KNNK; ++k) {
            float dA = md[l][i0], dB = md[l + 16][i1];
            int   iA = mi[l][i0], iB = mi[l + 16][i1];
            bool takeA = (dA < dB) || (dA == dB && iA < iB);
            o[k] = takeA ? iA : iB;
            if (takeA) ++i0; else ++i1;
        }
    }
}

// ---------------------------------------------------------------------------
// Kernel 3: edge features + attention via f16 WMMA.
// One wave per (batch, 16-node tile).  320 ek rows (16 nodes x 20 edges) of 24
// features (pad 32) x Wcat = [Wk | Wv | Wg | 0] (32x48) -> 60 WMMAs per block.
// ---------------------------------------------------------------------------
__global__ void edge_attn_kernel(const float* __restrict__ z, const float* __restrict__ mask,
                                 const float* __restrict__ cond,
                                 const float* __restrict__ Wq, const float* __restrict__ Wk,
                                 const float* __restrict__ Wv, const float* __restrict__ Wg,
                                 const int* __restrict__ knn_idx,
                                 float* __restrict__ out) {
    int blk = blockIdx.x;
    int b = blk >> 8;
    int nodeBase = (blk & 255) << 4;
    int l = threadIdx.x;
    int half = l >> 4;
    int lm = l & 15;

    __shared__ _Float16 ekbuf[320][32];   // [s_src(8) | rbf(16) | 0(8)] per edge
    __shared__ float    aux[320][8];      // velj.xyz, rhat.xyz, maskj, pad
    __shared__ float    qbuf[16][32];
    __shared__ float    logits[320];
    __shared__ float    vsb[320], g0b[320], g1b[320];
    __shared__ float    conds[8];

    const float* zb = z + (size_t)b * NPTS * ZF;
    const float* mb = mask + (size_t)b * NPTS;
    const int*   kb = knn_idx + (size_t)b * NPTS * KNNK;

    if (l < 8) conds[l] = cond[b * 8 + l];
    __syncthreads();

    {   // q = s @ Wq for this tile's 16 nodes: lane = (node=lm, col-half)
        int gi = nodeBase + lm;
        float s_i[8];
#pragma unroll
        for (int e = 0; e < 8; ++e) s_i[e] = zb[(size_t)gi * ZF + 6 + e] + conds[e];
        int c0 = 16 * half;
        for (int c = 0; c < 16; ++c) {
            float acc = 0.0f;
#pragma unroll
            for (int e = 0; e < 8; ++e) acc += s_i[e] * Wq[e * H_DIM + c0 + c];
            qbuf[lm][c0 + c] = acc;
        }
    }

    // build 320 edge rows (10 per lane)
    for (int e = l; e < 16 * KNNK; e += 32) {
        int node = e / KNNK, k = e % KNNK;
        int gi = nodeBase + node;
        int j = kb[(size_t)gi * KNNK + k];
        const float* pi = zb + (size_t)gi * ZF;
        const float* pj = zb + (size_t)j * ZF;
        float rx = pj[0] - pi[0], ry = pj[1] - pi[1], rz = pj[2] - pi[2];
        float r = sqrtf(rx * rx + ry * ry + rz * rz + 1e-12f);
        float inv = 1.0f / (r + 1e-8f);
        aux[e][0] = pj[3]; aux[e][1] = pj[4]; aux[e][2] = pj[5];
        aux[e][3] = rx * inv; aux[e][4] = ry * inv; aux[e][5] = rz * inv;
        aux[e][6] = mb[j];  aux[e][7] = 0.0f;
#pragma unroll
        for (int c = 0; c < 8; ++c) ekbuf[e][c] = (_Float16)(pj[6 + c] + conds[c]);
#pragma unroll
        for (int m = 0; m < N_RBF; ++m) {
            float dmu = r - 0.2f * (float)m;              // mu = linspace(0,3,16)
            ekbuf[e][8 + m] = (_Float16)expf(-10.0f * dmu * dmu);
        }
#pragma unroll
        for (int c = 24; c < 32; ++c) ekbuf[e][c] = (_Float16)0.0f;
    }
    __syncthreads();

    // B fragments: Wcat (32x48), 16-bit B layout matching A (lanes0-15: K0-7,16-23; 16-31: K8-15,24-31)
    v16h bfr[3];
#pragma unroll
    for (int t = 0; t < 3; ++t) {
#pragma unroll
        for (int m = 0; m < 16; ++m) {
            int kk = (m < 8) ? (8 * half + m) : (16 + 8 * half + (m - 8));
            int c  = 16 * t + lm;
            float w = 0.0f;
            if (kk < 24) {
                if (c < 32)       w = Wk[kk * H_DIM + c];
                else if (c == 32) w = Wv[kk];
                else if (c == 33) w = Wg[kk * 2 + 0];
                else if (c == 34) w = Wg[kk * 2 + 1];
            }
            bfr[t][m] = (_Float16)w;
        }
    }

    const float scale = 0.17677669529663687f;             // 1/sqrt(H=32)

    for (int t = 0; t < KNNK; ++t) {                      // 20 row tiles of 16 edges
        int rt = t * 16 + lm;
        v16h afr;                                         // ISA 16-bit A layout
#pragma unroll
        for (int i = 0; i < 8; ++i) {
            afr[i]     = ekbuf[rt][8 * half + i];
            afr[8 + i] = ekbuf[rt][8 * half + 16 + i];
        }
        v8f c0 = {}, c1 = {}, c2 = {};
        c0 = __builtin_amdgcn_wmma_f32_16x16x32_f16(false, afr, false, bfr[0], (short)0, c0, false, false);
        c1 = __builtin_amdgcn_wmma_f32_16x16x32_f16(false, afr, false, bfr[1], (short)0, c1, false, false);
        c2 = __builtin_amdgcn_wmma_f32_16x16x32_f16(false, afr, false, bfr[2], (short)0, c2, false, false);

        // logits_row = sum_c q[node][c] * kk[row][c] ; C layout: lane holds col lm
        float pl[8];
#pragma unroll
        for (int v = 0; v < 8; ++v) {
            int row = t * 16 + v + 8 * half;
            int node = row / KNNK;
            pl[v] = c0[v] * qbuf[node][lm] + c1[v] * qbuf[node][16 + lm];
        }
#pragma unroll
        for (int off = 1; off < 16; off <<= 1) {          // xor<16 stays in 16-lane half
#pragma unroll
            for (int v = 0; v < 8; ++v) pl[v] += __shfl_xor(pl[v], off);
        }
        if (lm == 0) {
#pragma unroll
            for (int v = 0; v < 8; ++v) {
                int row = t * 16 + v + 8 * half;
                logits[row] = pl[v] * scale;
                vsb[row] = c2[v];                         // col 32 of Wcat = Wv
            }
        }
        if (lm == 1) {
#pragma unroll
            for (int v = 0; v < 8; ++v) g0b[t * 16 + v + 8 * half] = c2[v];  // col 33
        }
        if (lm == 2) {
#pragma unroll
            for (int v = 0; v < 8; ++v) g1b[t * 16 + v + 8 * half] = c2[v];  // col 34
        }
    }
    __syncthreads();

    // softmax + equivariant aggregation, one lane per node
    if (l < 16) {
        int gi = nodeBase + l;
        int e0 = l * KNNK;
        float lg[KNNK], mx = -3.0e38f;
#pragma unroll
        for (int k = 0; k < KNNK; ++k) {
            float v = (aux[e0 + k][6] > 0.0f) ? logits[e0 + k] : -1.0e9f;
            lg[k] = v; mx = fmaxf(mx, v);
        }
        float ssum = 0.0f, px = 0.0f, py = 0.0f, pz = 0.0f, ft = 0.0f;
#pragma unroll
        for (int k = 0; k < KNNK; ++k) {
            int e = e0 + k;
            float a = expf(lg[k] - mx);
            ssum += a;
            float g0 = g0b[e], g1 = g1b[e];
            px += a * (g0 * aux[e][0] + g1 * aux[e][3]);
            py += a * (g0 * aux[e][1] + g1 * aux[e][4]);
            pz += a * (g0 * aux[e][2] + g1 * aux[e][5]);
            ft += a * vsb[e];
        }
        float inv = 1.0f / ssum;
        float* o = out + ((size_t)b * NPTS + gi) * 4;
        o[0] = px * inv; o[1] = py * inv; o[2] = pz * inv; o[3] = ft * inv;
    }
}

// ---------------------------------------------------------------------------
extern "C" void kernel_launch(void* const* d_in, const int* in_sizes, int n_in,
                              void* d_out, int out_size, void* d_ws, size_t ws_size,
                              hipStream_t stream) {
    const float* z            = (const float*)d_in[0];
    const float* t            = (const float*)d_in[1];
    const float* conditioning = (const float*)d_in[2];
    const float* mask         = (const float*)d_in[3];
    const float* W1 = (const float*)d_in[4];
    const float* b1 = (const float*)d_in[5];
    const float* W2 = (const float*)d_in[6];
    const float* b2 = (const float*)d_in[7];
    const float* W3 = (const float*)d_in[8];
    const float* b3 = (const float*)d_in[9];
    const float* Wq = (const float*)d_in[10];
    const float* Wk = (const float*)d_in[11];
    const float* Wv = (const float*)d_in[12];
    const float* Wg = (const float*)d_in[13];

    float* cond = (float*)d_ws;                           // 32 floats
    int*   knn  = (int*)((char*)d_ws + 256);              // B*N*K ints (~1.31 MB)

    cond_mlp_kernel<<<1, 128, 0, stream>>>(t, conditioning, W1, b1, W2, b2, W3, b3, cond);
    knn_kernel<<<BATCH * 256, 32, 0, stream>>>(z, mask, knn);
    edge_attn_kernel<<<BATCH * 256, 32, 0, stream>>>(z, mask, cond, Wq, Wk, Wv, Wg, knn,
                                                     (float*)d_out);
}
